// CollaborativeAttention_75256416960845
// MI455X (gfx1250) — matrix operand order repaired and checked by reading, not compiled, not run on hardware
//
#include <hip/hip_runtime.h>

// ---------------- problem constants ----------------
#define NBATCH 8
#define SEQ    512
#define DMODEL 1024
#define NHEAD  16
#define DHEAD  64
#define BS_TOK (NBATCH * SEQ)          // 4096 rows
#define ATT_SCALE 0.125f               // 1/sqrt(64)

// ---------------- types ----------------
typedef __bf16 bf16_t;
typedef __attribute__((ext_vector_type(16))) __bf16 v16bf;
typedef __attribute__((ext_vector_type(8)))  float  v8f;

struct alignas(16) u128 { unsigned int x, y, z, w; };
struct alignas(16) f128 { float x, y, z, w; };

__device__ __forceinline__ bf16_t f2bf(float f) {
  unsigned u = __builtin_bit_cast(unsigned, f);
  u += 0x7FFFu + ((u >> 16) & 1u);               // round-to-nearest-even
  unsigned short h = (unsigned short)(u >> 16);
  return __builtin_bit_cast(bf16_t, h);
}

__device__ __forceinline__ v8f wmma_bf16(v16bf a, v16bf b, v8f c) {
  // D = A(16x32 bf16) * B(32x16 bf16) + C(16x16 f32)
  return __builtin_amdgcn_wmma_f32_16x16x32_bf16(false, a, false, b,
                                                 (short)0, c, false, false);
}

// ---------------- async global->LDS copy (CDNA5 ASYNCcnt path) ----------------
#if defined(__AMDGCN__) && __has_builtin(__builtin_amdgcn_global_load_async_to_lds_b128)
#define HAVE_ASYNC_LDS 1
#else
#define HAVE_ASYNC_LDS 0
#endif

typedef int i32x4 __attribute__((vector_size(16)));          // matches builtin proto
typedef __attribute__((address_space(1))) i32x4 gi32x4_t;    // global int4
typedef __attribute__((address_space(3))) i32x4 li32x4_t;    // LDS int4

// copy 16 bytes global -> LDS
__device__ __forceinline__ void cp16(const bf16_t* g, bf16_t* l) {
#if HAVE_ASYNC_LDS
  __builtin_amdgcn_global_load_async_to_lds_b128(
      (gi32x4_t*)(unsigned long long)g,
      (li32x4_t*)(unsigned)(unsigned long long)l,
      0, 0);
#else
  *(u128*)l = *(const u128*)g;
#endif
}

__device__ __forceinline__ void wait_async0() {
#if HAVE_ASYNC_LDS
#if __has_builtin(__builtin_amdgcn_s_wait_asynccnt)
  __builtin_amdgcn_s_wait_asynccnt(0);
#else
  asm volatile("s_wait_asynccnt 0" ::: "memory");
#endif
#endif
}

union frag_u { u128 u[2]; v16bf v; };

// A-fragment (16x32, M x K), source row-major [row][k], ld in elements.
// lanes 0-15: row=lane,   K in {0..7}  and {16..23}
// lanes16-31: row=lane-16,K in {8..15} and {24..31}
__device__ __forceinline__ v16bf ld_frag_a(const bf16_t* base, int ld,
                                           int row0, int k0, int lane) {
  int r  = row0 + (lane & 15);
  int kb = (lane >> 4) << 3;                     // 0 or 8
  const bf16_t* p = base + (size_t)r * ld + k0 + kb;
  frag_u f;
  f.u[0] = *(const u128*)(p);
  f.u[1] = *(const u128*)(p + 16);
  return f.v;
}

// B-fragment (32x16, K x N), source staged K-contiguous per column:
// stage[col][k].  lanes 0-15: col=lane, K in {0..7} and {16..23}; lanes16-31: col=lane-16, K in {8..15} and {24..31}
__device__ __forceinline__ v16bf ld_frag_b(const bf16_t* base, int ld,
                                           int col0, int k0, int lane) {
  int c  = col0 + (lane & 15);
  int kb = (lane >> 4) << 3;                     // 0 or 8
  const bf16_t* p = base + (size_t)c * ld + k0 + kb;
  frag_u f;
  f.u[0] = *(const u128*)(p);
  f.u[1] = *(const u128*)(p + 16);
  return f.v;
}

// ---------------- f32 -> bf16 convert (activations) ----------------
__global__ void __launch_bounds__(256)
k_cvt_bf16(const float* __restrict__ in, bf16_t* __restrict__ out, int n4) {
  int i = blockIdx.x * 256 + threadIdx.x;
  if (i < n4) {
    f128 v = ((const f128*)in)[i];
    out[i * 4 + 0] = f2bf(v.x);
    out[i * 4 + 1] = f2bf(v.y);
    out[i * 4 + 2] = f2bf(v.z);
    out[i * 4 + 3] = f2bf(v.w);
  }
}

// ---------------- f32 -> bf16 convert + transpose (weights) ----------------
// in: [K][N] f32 row-major  ->  out: [N][K] bf16 row-major
__global__ void __launch_bounds__(256)
k_cvt_t(const float* __restrict__ in, bf16_t* __restrict__ out, int K, int N) {
  __shared__ float t[32][33];
  const int bx = blockIdx.x * 32;                // N tile
  const int by = blockIdx.y * 32;                // K tile
  const int tx = threadIdx.x & 31;
  const int ty = threadIdx.x >> 5;               // 0..7
  #pragma unroll
  for (int i = 0; i < 32; i += 8)
    t[ty + i][tx] = in[(size_t)(by + ty + i) * N + bx + tx];
  __syncthreads();
  #pragma unroll
  for (int i = 0; i < 32; i += 8)
    out[(size_t)(bx + ty + i) * K + by + tx] = f2bf(t[tx][ty + i]);
}

// ---------------- WMMA GEMM: C = A @ W + bias (W pre-transposed) ----------------
// A [M][kSplit] (A2 [M][K-kSplit] virtual concat along K), Wt [N][K] row-major.
// EPI: 0 = bias -> bf16 out; 1 = bias -> f32 + bf16 out; 2 = sigmoid(bias+) -> f32
#define BM 128
#define BN 128
#define BK 32
#define LDA (BK + 8)
#define LDB (BK + 8)

template<int EPI>
__global__ void __launch_bounds__(256)
k_gemm_bf16(const bf16_t* __restrict__ A, const bf16_t* __restrict__ A2, int kSplit,
            const bf16_t* __restrict__ Wt, const float* __restrict__ bias,
            float* __restrict__ outF, bf16_t* __restrict__ outB,
            int M, int N, int K)
{
  __shared__ __attribute__((aligned(16))) bf16_t sA [2][BM * LDA];   // 2 x 10 KB
  __shared__ __attribute__((aligned(16))) bf16_t sBt[2][BN * LDB];   // 2 x 10 KB

  const int tid  = threadIdx.x;
  const int lane = tid & 31;
  const int wave = tid >> 5;
  const int bm = blockIdx.y * BM;
  const int bn = blockIdx.x * BN;
  const int wm = (wave >> 1) * 32;               // 4 waves along M
  const int wn = (wave &  1) * 64;               // 2 waves along N

  const v8f zero = {0.f,0.f,0.f,0.f,0.f,0.f,0.f,0.f};
  v8f acc[2][4];
  #pragma unroll
  for (int mi = 0; mi < 2; ++mi)
    #pragma unroll
    for (int ni = 0; ni < 4; ++ni) acc[mi][ni] = zero;

  const int sr = tid >> 1;                       // staging row 0..127 (A and Wt)
  const int ss = (tid & 1) * 16;                 // staging k-seg 0/16

  // stage one (A,B) tile pair into LDS buffer `buf` (async when available)
  auto stage = [&](int buf, int k0) {
    int gk = k0 + ss;
    const bf16_t* asrc = (gk < kSplit)
        ? (A  + (size_t)(bm + sr) * kSplit + gk)
        : (A2 + (size_t)(bm + sr) * kSplit + (gk - kSplit));
    bf16_t* adst = &sA[buf][sr * LDA + ss];
    cp16(asrc,     adst);
    cp16(asrc + 8, adst + 8);
    const bf16_t* wsrc = Wt + (size_t)(bn + sr) * K + k0 + ss;
    bf16_t* wdst = &sBt[buf][sr * LDB + ss];
    cp16(wsrc,     wdst);
    cp16(wsrc + 8, wdst + 8);
  };

  stage(0, 0);
  wait_async0();
  __syncthreads();

  int cur = 0;
  for (int k0 = 0; k0 < K; k0 += BK) {
    if (k0 + BK < K) stage(cur ^ 1, k0 + BK);    // pipeline next tile
    if (k0 + 2 * BK < K)                         // prefetch tile after next
      __builtin_prefetch(Wt + (size_t)(bn + sr) * K + k0 + 2 * BK + ss, 0, 1);

    v16bf bfr[4];
    #pragma unroll
    for (int ni = 0; ni < 4; ++ni)
      bfr[ni] = ld_frag_b(&sBt[cur][0], LDB, wn + ni * 16, 0, lane);
    #pragma unroll
    for (int mi = 0; mi < 2; ++mi) {
      v16bf af = ld_frag_a(&sA[cur][0], LDA, wm + mi * 16, 0, lane);
      #pragma unroll
      for (int ni = 0; ni < 4; ++ni)
        acc[mi][ni] = wmma_bf16(af, bfr[ni], acc[mi][ni]);
    }
    wait_async0();                               // next tile landed in LDS
    __syncthreads();                             // all waves done with `cur`
    cur ^= 1;
  }

  // epilogue — C layout: lane&15 = n, (lane>>4)*8 + vgpr = m
  const int nl = lane & 15;
  const int mh = (lane >> 4) * 8;
  #pragma unroll
  for (int mi = 0; mi < 2; ++mi) {
    #pragma unroll
    for (int ni = 0; ni < 4; ++ni) {
      int col = bn + wn + ni * 16 + nl;
      float bv = bias[col];
      #pragma unroll
      for (int v = 0; v < 8; ++v) {
        int row = bm + wm + mi * 16 + mh + v;
        size_t o = (size_t)row * N + col;
        float val = acc[mi][ni][v] + bv;
        if (EPI == 0) {
          outB[o] = f2bf(val);
        } else if (EPI == 1) {
          outF[o] = val;
          outB[o] = f2bf(val);
        } else {
          outF[o] = 1.f / (1.f + __expf(-val));
        }
      }
    }
  }
}

// ---------------- attention: softmax(Q K^T * s) V  (per b,h, 32 q-rows) --------
// The reference's GAMMA*log(row_mean+eps) term is constant along the softmax
// axis -> softmax-invariant -> mathematically a no-op; skipped.
#define LDP 528
#define LDQ 72
#define LDK 72
#define LDV 40
#define SC_BYTES (32 * 512 * 4)                  // 65536
#define PR_BYTES (32 * LDP * 2)                  // 33792
#define QT_BYTES (32 * LDQ * 2)                  // 4608
#define KT_BYTES (64 * LDK * 2)                  // 9216 (reused as transposed V)
#define ATTN_SMEM (SC_BYTES + PR_BYTES + QT_BYTES + KT_BYTES)

__global__ void __launch_bounds__(256)
k_attn(const bf16_t* __restrict__ Q, const bf16_t* __restrict__ Kb,
       const bf16_t* __restrict__ Vb, bf16_t* __restrict__ ctx)
{
  extern __shared__ __attribute__((aligned(16))) char smem[];
  float*  sc = (float*)smem;                                     // [32][512] scores
  bf16_t* pr = (bf16_t*)(smem + SC_BYTES);                       // [32][528] probs
  bf16_t* qt = (bf16_t*)(smem + SC_BYTES + PR_BYTES);            // [32][72]  Q tile
  bf16_t* kt = (bf16_t*)(smem + SC_BYTES + PR_BYTES + QT_BYTES); // [64][72]  K / V^T

  const int tid = threadIdx.x, lane = tid & 31, wave = tid >> 5;
  const int bh = blockIdx.x >> 4;                // 16 q-blocks per (b,h)
  const int qb = blockIdx.x & 15;
  const int b  = bh >> 4;                        // NHEAD = 16
  const int h  = bh & 15;
  const int rowbase = b * SEQ + qb * 32;
  const int colbase = h * DHEAD;
  const int mi = wave >> 2, ni = wave & 3;       // wave -> one 16x16 tile of 32x64

  const v8f zero = {0.f,0.f,0.f,0.f,0.f,0.f,0.f,0.f};

  { // stage Q tile 32x64 (async)
    int r = tid >> 3, cs = (tid & 7) * 8;
    cp16(Q + (size_t)(rowbase + r) * DMODEL + colbase + cs, &qt[r * LDQ + cs]);
  }

  // ---- phase 1: scores = Q K^T * scale, strip of 32 rows x 512 keys ----
  for (int kb = 0; kb < 8; ++kb) {
    __syncthreads();                             // readers of kt from prev round done
    { // stage K tile 64 keys x 64 dims (rows = key index -> already "B transposed")
      int r = tid >> 2, cs = (tid & 3) * 16;
      const bf16_t* p = Kb + (size_t)(b * SEQ + kb * 64 + r) * DMODEL + colbase + cs;
      bf16_t* d = &kt[r * LDK + cs];
      cp16(p,     d);
      cp16(p + 8, d + 8);
    }
    wait_async0();
    __syncthreads();
    v8f acc = zero;
    #pragma unroll
    for (int ks = 0; ks < 2; ++ks) {             // DH=64 -> two K=32 WMMA steps
      v16bf af  = ld_frag_a(qt, LDQ, mi * 16, ks * 32, lane);
      v16bf bfv = ld_frag_b(kt, LDK, ni * 16, ks * 32, lane);
      acc = wmma_bf16(af, bfv, acc);
    }
    int n = lane & 15, mh = (lane >> 4) * 8;
    #pragma unroll
    for (int v = 0; v < 8; ++v)
      sc[(mi * 16 + mh + v) * 512 + kb * 64 + ni * 16 + n] = acc[v] * ATT_SCALE;
  }
  __syncthreads();

  // ---- phase 2: row softmax, emit bf16 probabilities ----
  if (tid < 32) {
    float* row = sc + tid * 512;
    float mx = -3.4e38f;
    for (int j = 0; j < 512; ++j) mx = fmaxf(mx, row[j]);
    float s = 0.f;
    for (int j = 0; j < 512; ++j) { float e = __expf(row[j] - mx); row[j] = e; s += e; }
    float inv = 1.f / s;
    bf16_t* pro = pr + tid * LDP;
    for (int j = 0; j < 512; ++j) pro[j] = f2bf(row[j] * inv);
  }

  // ---- phase 3: O = P V ----
  v8f oacc = zero;
  for (int kk = 0; kk < 16; ++kk) {              // 512 keys in K=32 steps
    __syncthreads();
    { // stage V block 32x64 transposed -> vtT[n][k] (reuse kt space)
      int r = tid >> 3, cs = (tid & 7) * 8;
      const bf16_t* p = Vb + (size_t)(b * SEQ + kk * 32 + r) * DMODEL + colbase + cs;
      union { u128 u; unsigned short s8[8]; } uu;
      uu.u = *(const u128*)p;
      #pragma unroll
      for (int j = 0; j < 8; ++j)
        kt[(cs + j) * LDV + r] = __builtin_bit_cast(bf16_t, uu.s8[j]);
    }
    __syncthreads();
    v16bf af  = ld_frag_a(pr, LDP, mi * 16, kk * 32, lane);
    v16bf bfv = ld_frag_b(kt, LDV, ni * 16, 0, lane);
    oacc = wmma_bf16(af, bfv, oacc);
  }

  { // write ctx (bf16) for the output projection GEMM
    int n = lane & 15, mh = (lane >> 4) * 8;
    #pragma unroll
    for (int v = 0; v < 8; ++v)
      ctx[(size_t)(rowbase + mi * 16 + mh + v) * DMODEL + colbase + ni * 16 + n]
          = f2bf(oacc[v]);
  }
}

// ---------------- fused gate-mix + LayerNorm ----------------
__global__ void __launch_bounds__(256)
k_fuse_ln(const float* __restrict__ x, const float* __restrict__ a,
          const float* __restrict__ g, const float* __restrict__ lng,
          const float* __restrict__ lnb, float* __restrict__ out)
{
  __shared__ float rs[256], rs2[256];
  const int r = blockIdx.x, tid = threadIdx.x;
  const float* xp = x + (size_t)r * DMODEL;
  const float* ap = a + (size_t)r * DMODEL;
  const float* gp = g + (size_t)r * DMODEL;
  float y[4]; float s = 0.f, s2 = 0.f;
  #pragma unroll
  for (int i = 0; i < 4; ++i) {
    int c = tid + i * 256;
    float gv = gp[c];
    float yv = gv * xp[c] + (1.f - gv) * ap[c];
    y[i] = yv; s += yv; s2 += yv * yv;
  }
  rs[tid] = s; rs2[tid] = s2;
  __syncthreads();
  for (int o = 128; o > 0; o >>= 1) {
    if (tid < o) { rs[tid] += rs[tid + o]; rs2[tid] += rs2[tid + o]; }
    __syncthreads();
  }
  float mu  = rs[0]  * (1.f / DMODEL);
  float var = rs2[0] * (1.f / DMODEL) - mu * mu;
  float inv = rsqrtf(var + 1e-5f);
  float* op = out + (size_t)r * DMODEL;
  #pragma unroll
  for (int i = 0; i < 4; ++i) {
    int c = tid + i * 256;
    op[c] = lng[c] * (y[i] - mu) * inv + lnb[c];
  }
}

// ---------------- host orchestration ----------------
extern "C" void kernel_launch(void* const* d_in, const int* in_sizes, int n_in,
                              void* d_out, int out_size, void* d_ws, size_t ws_size,
                              hipStream_t stream)
{
  (void)in_sizes; (void)n_in; (void)out_size; (void)ws_size;

  const float* title   = (const float*)d_in[0];
  const float* content = (const float*)d_in[1];
  // weights in dict order: t2c {wq,bq,wk,bk,wv,bv,wo,bo} then c2t, gate, ln
  const float* w_f32[8] = {
    (const float*)d_in[2],  (const float*)d_in[4],  (const float*)d_in[6],  (const float*)d_in[8],
    (const float*)d_in[10], (const float*)d_in[12], (const float*)d_in[14], (const float*)d_in[16]
  };
  const float* bias_q1 = (const float*)d_in[3];
  const float* bias_k1 = (const float*)d_in[5];
  const float* bias_v1 = (const float*)d_in[7];
  const float* bias_o1 = (const float*)d_in[9];
  const float* bias_q2 = (const float*)d_in[11];
  const float* bias_k2 = (const float*)d_in[13];
  const float* bias_v2 = (const float*)d_in[15];
  const float* bias_o2 = (const float*)d_in[17];
  const float* gate_w  = (const float*)d_in[18];
  const float* gate_b  = (const float*)d_in[19];
  const float* ln_g    = (const float*)d_in[20];
  const float* ln_b    = (const float*)d_in[21];

  // workspace carve
  char* wp = (char*)d_ws;
  auto alloc = [&](size_t bytes) -> char* {
    char* p = wp; wp += (bytes + 255) & ~(size_t)255; return p;
  };
  const size_t NB = (size_t)BS_TOK * DMODEL;     // 4M elements
  bf16_t* titlb = (bf16_t*)alloc(NB * 2);
  bf16_t* contb = (bf16_t*)alloc(NB * 2);
  bf16_t* wb[8];                                 // transposed bf16 weights [N][K]
  for (int i = 0; i < 8; ++i) wb[i] = (bf16_t*)alloc((size_t)DMODEL * DMODEL * 2);
  bf16_t* gwb  = (bf16_t*)alloc((size_t)2 * DMODEL * DMODEL * 2);  // [1024][2048]
  bf16_t* q1   = (bf16_t*)alloc(NB * 2);
  bf16_t* k1   = (bf16_t*)alloc(NB * 2);
  bf16_t* v1   = (bf16_t*)alloc(NB * 2);
  bf16_t* q2   = (bf16_t*)alloc(NB * 2);
  bf16_t* k2   = (bf16_t*)alloc(NB * 2);
  bf16_t* v2   = (bf16_t*)alloc(NB * 2);
  bf16_t* ctx1 = (bf16_t*)alloc(NB * 2);
  bf16_t* ctx2 = (bf16_t*)alloc(NB * 2);
  bf16_t* aB1  = (bf16_t*)alloc(NB * 2);
  bf16_t* aB2  = (bf16_t*)alloc(NB * 2);
  float*  aF1  = (float*)alloc(NB * 4);
  float*  aF2  = (float*)alloc(NB * 4);
  float*  gT   = (float*)alloc(NB * 4);
  float*  gC   = (float*)alloc(NB * 4);

  // ---- converts ----
  {
    int n4 = (int)(NB / 4);
    dim3 cg((n4 + 255) / 256);
    k_cvt_bf16<<<cg, 256, 0, stream>>>(title,   titlb, n4);
    k_cvt_bf16<<<cg, 256, 0, stream>>>(content, contb, n4);
    dim3 wg(DMODEL / 32, DMODEL / 32);           // (N/32, K/32)
    for (int i = 0; i < 8; ++i)
      k_cvt_t<<<wg, 256, 0, stream>>>(w_f32[i], wb[i], DMODEL, DMODEL);
    dim3 gg(DMODEL / 32, 2 * DMODEL / 32);       // gate: K=2048, N=1024
    k_cvt_t<<<gg, 256, 0, stream>>>(gate_w, gwb, 2 * DMODEL, DMODEL);
  }

  dim3 gemmGrid(DMODEL / BN, BS_TOK / BM);       // (8, 32)

  // ---- QKV projections (t2c: Q from content, K/V from title; c2t swapped) ----
  k_gemm_bf16<0><<<gemmGrid, 256, 0, stream>>>(contb, contb, DMODEL, wb[0], bias_q1, nullptr, q1, BS_TOK, DMODEL, DMODEL);
  k_gemm_bf16<0><<<gemmGrid, 256, 0, stream>>>(titlb, titlb, DMODEL, wb[1], bias_k1, nullptr, k1, BS_TOK, DMODEL, DMODEL);
  k_gemm_bf16<0><<<gemmGrid, 256, 0, stream>>>(titlb, titlb, DMODEL, wb[2], bias_v1, nullptr, v1, BS_TOK, DMODEL, DMODEL);
  k_gemm_bf16<0><<<gemmGrid, 256, 0, stream>>>(titlb, titlb, DMODEL, wb[4], bias_q2, nullptr, q2, BS_TOK, DMODEL, DMODEL);
  k_gemm_bf16<0><<<gemmGrid, 256, 0, stream>>>(contb, contb, DMODEL, wb[5], bias_k2, nullptr, k2, BS_TOK, DMODEL, DMODEL);
  k_gemm_bf16<0><<<gemmGrid, 256, 0, stream>>>(contb, contb, DMODEL, wb[6], bias_v2, nullptr, v2, BS_TOK, DMODEL, DMODEL);

  // ---- attention ----
  dim3 attnGrid(NBATCH * NHEAD * (SEQ / 32));    // 2048 workgroups
  k_attn<<<attnGrid, 256, ATTN_SMEM, stream>>>(q1, k1, v1, ctx1);
  k_attn<<<attnGrid, 256, ATTN_SMEM, stream>>>(q2, k2, v2, ctx2);

  // ---- output projections (f32 for fuse + bf16 copy for gate concat) ----
  k_gemm_bf16<1><<<gemmGrid, 256, 0, stream>>>(ctx1, ctx1, DMODEL, wb[3], bias_o1, aF1, aB1, BS_TOK, DMODEL, DMODEL);
  k_gemm_bf16<1><<<gemmGrid, 256, 0, stream>>>(ctx2, ctx2, DMODEL, wb[7], bias_o2, aF2, aB2, BS_TOK, DMODEL, DMODEL);

  // ---- gates: sigmoid([x || attn] @ gate_w + b), concat virtualized via A2 ----
  k_gemm_bf16<2><<<gemmGrid, 256, 0, stream>>>(titlb, aB1, DMODEL, gwb, gate_b, gT, nullptr, BS_TOK, DMODEL, 2 * DMODEL);
  k_gemm_bf16<2><<<gemmGrid, 256, 0, stream>>>(contb, aB2, DMODEL, gwb, gate_b, gC, nullptr, BS_TOK, DMODEL, 2 * DMODEL);

  // ---- fused gate mix + LayerNorm -> outputs ----
  float* outF = (float*)d_out;
  k_fuse_ln<<<dim3(BS_TOK), 256, 0, stream>>>(title,   aF1, gT, ln_g, ln_b, outF);
  k_fuse_ln<<<dim3(BS_TOK), 256, 0, stream>>>(content, aF2, gC, ln_g, ln_b, outF + NB);
}
